// CapsNet_19877108646183
// MI455X (gfx1250) — matrix operand order repaired and checked by reading, not compiled, not run on hardware
//
#include <hip/hip_runtime.h>
#include <hip/hip_bf16.h>

typedef __attribute__((ext_vector_type(16))) _Float16 v16h;
typedef __attribute__((ext_vector_type(8)))  float    v8f;
typedef __attribute__((vector_size(16)))     int      vi4;   // matches builtin's V4i param

#define GM 256        // conv2 out channels
#define GK 20736      // 256 * 81
#define GN 18432      // 512 * 36
#define NPIX1 400     // 20*20
#define NB 512

union Frag16 { uint4 u4[2]; v16h h; };

// ---- CDNA5 async global->LDS copy (ASYNCcnt-tracked), with safe fallback ----
#if defined(__has_builtin)
#if __has_builtin(__builtin_amdgcn_global_load_async_to_lds_b128)
#define HAVE_ASYNC_LDS 1
#endif
#endif

#ifdef HAVE_ASYNC_LDS
__device__ __forceinline__ void async_copy_b128(const _Float16* g, _Float16* l) {
    typedef __attribute__((address_space(1))) vi4 gvi4;   // global
    typedef __attribute__((address_space(3))) vi4 lvi4;   // LDS
    __builtin_amdgcn_global_load_async_to_lds_b128(
        (gvi4*)(vi4*)const_cast<_Float16*>(g),
        (lvi4*)(vi4*)l,
        /*offset=*/0, /*cpol=*/0);
}
__device__ __forceinline__ void wait_async_all() {
#if __has_builtin(__builtin_amdgcn_s_wait_asynccnt)
    __builtin_amdgcn_s_wait_asynccnt(0);
#else
    asm volatile("s_wait_asynccnt 0" ::: "memory");
#endif
}
#else
__device__ __forceinline__ void async_copy_b128(const _Float16* g, _Float16* l) {
    *(uint4*)l = *(const uint4*)g;
}
__device__ __forceinline__ void wait_async_all() {}
#endif

// ---------------- conv2 weight pack: fp32 (oc,cin,9,9) -> f16 [oc][GK] ----------------
__global__ __launch_bounds__(256) void packw2_kernel(const float* __restrict__ w,
                                                     _Float16* __restrict__ A2) {
    size_t i = (size_t)blockIdx.x * 256 + threadIdx.x;
    if (i < (size_t)GM * GK) A2[i] = (_Float16)w[i];
}

// ---------------- conv1: (512,1,28,28) -> (512,256,20,20) fp32 (no bias; BN cancels it) ----
__global__ __launch_bounds__(256) void conv1_kernel(const float* __restrict__ x,
                                                    const float* __restrict__ w,
                                                    float* __restrict__ y) {
    const int pix = blockIdx.x;                 // n*400 + oh*20 + ow
    const int n = pix / NPIX1, p = pix % NPIX1;
    const int oh = p / 20, ow = p % 20;
    __shared__ float patch[81];
    if (threadIdx.x < 81) {
        int kh = threadIdx.x / 9, kw = threadIdx.x % 9;
        patch[threadIdx.x] = x[(size_t)n * 784 + (size_t)(oh + kh) * 28 + (ow + kw)];
    }
    __syncthreads();
    const int c = threadIdx.x;
    const float* wc = w + (size_t)c * 81;
    float s = 0.f;
    #pragma unroll
    for (int k = 0; k < 81; ++k) s = fmaf(patch[k], wc[k], s);
    y[((size_t)n * GM + c) * NPIX1 + p] = s;
}

// ---------------- BN1 stats -> fused scale/shift per channel ----------------
__global__ __launch_bounds__(256) void bnstat1_kernel(const float* __restrict__ y,
                                                      const float* __restrict__ g,
                                                      const float* __restrict__ b,
                                                      float* __restrict__ scale,
                                                      float* __restrict__ shift) {
    const int c = blockIdx.x;
    const int tid = threadIdx.x;
    float s = 0.f, ss = 0.f;
    for (int idx = tid; idx < NB * NPIX1; idx += 256) {
        int n = idx / NPIX1, p = idx % NPIX1;
        float v = y[((size_t)n * GM + c) * NPIX1 + p];
        s += v; ss += v * v;
    }
    __shared__ float rs[256], rq[256];
    rs[tid] = s; rq[tid] = ss;
    __syncthreads();
    for (int o = 128; o > 0; o >>= 1) {
        if (tid < o) { rs[tid] += rs[tid + o]; rq[tid] += rq[tid + o]; }
        __syncthreads();
    }
    if (tid == 0) {
        const float cnt = (float)(NB * NPIX1);
        float m = rs[0] / cnt;
        float var = rq[0] / cnt - m * m;
        float rinv = rsqrtf(var + 1e-5f);
        scale[c] = g[c] * rinv;
        shift[c] = b[c] - m * g[c] * rinv;
    }
}

// ---------------- BN1 apply + ReLU -> f16 activations ----------------
__global__ __launch_bounds__(256) void bnapply1_kernel(const float* __restrict__ y,
                                                       const float* __restrict__ scale,
                                                       const float* __restrict__ shift,
                                                       _Float16* __restrict__ yh) {
    size_t i = (size_t)blockIdx.x * 256 + threadIdx.x;
    if (i >= (size_t)NB * GM * NPIX1) return;
    int c = (int)((i / NPIX1) % GM);
    float v = fmaf(y[i], scale[c], shift[c]);
    yh[i] = (_Float16)fmaxf(v, 0.f);
}

// ---------------- im2col (stride-2 9x9) -> B2 f16 [j=GN][k=GK]  (K-major rows) ----------
__global__ __launch_bounds__(256) void im2col_kernel(const _Float16* __restrict__ yh,
                                                     _Float16* __restrict__ B2) {
    const int j = blockIdx.x;                   // n*36 + oh*6 + ow
    const int n = j / 36, p = j % 36;
    const int oh = p / 6, ow = p % 6;
    const _Float16* base = yh + (size_t)n * GM * NPIX1;
    _Float16* out = B2 + (size_t)j * GK;
    for (int k = threadIdx.x; k < GK; k += 256) {
        int cin = k / 81, r = k % 81;
        int kh = r / 9, kw = r % 9;
        out[k] = base[(size_t)cin * NPIX1 + (2 * oh + kh) * 20 + (2 * ow + kw)];
    }
}

// ---------------- conv2 as WMMA GEMM: C[GM][GN] = A2[GM][GK] * B2[GN][GK]^T ----------------
// BM=256 (full M), BN=64 per WG; 8 waves: 4 along M (64 rows) x 2 along N (32 cols).
// Each wave: 4 M-frags x 2 N-frags of v_wmma_f32_16x16x32_f16, f32 accumulate.
// Tiles are fetched with GLOBAL_LOAD_ASYNC_TO_LDS_B128 (ASYNCcnt) when available,
// overlapping the buf^1 DMA with WMMA compute on buf; one s_wait_asynccnt 0 + barrier.
__global__ __launch_bounds__(256) void conv2_gemm_kernel(const _Float16* __restrict__ A,
                                                         const _Float16* __restrict__ B,
                                                         float* __restrict__ C) {
    __shared__ alignas(16) _Float16 As[2][GM * 32];   // [m][k], k-contiguous
    __shared__ alignas(16) _Float16 Bs[2][64 * 32];   // [n][k], k-contiguous
    const int tid  = threadIdx.x;
    const int lane = tid & 31;
    const int wave = tid >> 5;
    const int wm = wave & 3;          // 0..3 -> 64 rows each
    const int wn = wave >> 2;         // 0..1 -> 32 cols each
    const int ml = lane & 15;
    const int half = lane >> 4;
    const size_t jbase = (size_t)blockIdx.x * 64;

    const v8f z8 = {0.f, 0.f, 0.f, 0.f, 0.f, 0.f, 0.f, 0.f};
    v8f acc[4][2];
    #pragma unroll
    for (int m = 0; m < 4; ++m)
        #pragma unroll
        for (int n = 0; n < 2; ++n) acc[m][n] = z8;

    auto load_tile = [&](int kt, int bb) {
        const int kb = kt * 32;
        #pragma unroll
        for (int it = 0; it < 4; ++it) {        // A tile: 256x32 f16 = 16KB
            int cc = tid + 256 * it;            // 0..1023 chunks of 16B
            int row = cc >> 2, seg = (cc & 3) * 8;
            async_copy_b128(A + (size_t)row * GK + kb + seg, &As[bb][row * 32 + seg]);
        }
        {                                        // B tile: 64x32 f16 = 4KB
            int row = tid >> 2, seg = (tid & 3) * 8;
            async_copy_b128(B + (jbase + row) * GK + kb + seg, &Bs[bb][row * 32 + seg]);
        }
    };

    load_tile(0, 0);
    wait_async_all();
    __syncthreads();

    int buf = 0;
    for (int kt = 0; kt < GK / 32; ++kt) {
        if (kt + 1 < GK / 32) load_tile(kt + 1, buf ^ 1);

        // A fragment (16x32 f16): two b128 per lane at elem offsets half*8 and 16+half*8
        Frag16 afr[4];
        #pragma unroll
        for (int m = 0; m < 4; ++m) {
            const _Float16* p = &As[buf][(size_t)(wm * 64 + m * 16 + ml) * 32];
            afr[m].u4[0] = *(const uint4*)(p + half * 8);
            afr[m].u4[1] = *(const uint4*)(p + 16 + half * 8);
        }
        // B fragment (32x16 f16, stored [n][k]): two b128 at offsets half*8 and 16+half*8
        Frag16 bfr[2];
        #pragma unroll
        for (int n = 0; n < 2; ++n) {
            const _Float16* p = &Bs[buf][(size_t)(wn * 32 + n * 16 + ml) * 32];
            bfr[n].u4[0] = *(const uint4*)(p + half * 8);
            bfr[n].u4[1] = *(const uint4*)(p + 16 + half * 8);
        }
        #pragma unroll
        for (int m = 0; m < 4; ++m)
            #pragma unroll
            for (int n = 0; n < 2; ++n)
                acc[m][n] = __builtin_amdgcn_wmma_f32_16x16x32_f16(
                    false, afr[m].h, false, bfr[n].h,
                    (short)0, acc[m][n], false, false);

        wait_async_all();
        __syncthreads();
        buf ^= 1;
    }

    // Epilogue: C/D layout -> VGPR r: lanes 0-15 M=r, lanes 16-31 M=r+8; N=lane%16
    #pragma unroll
    for (int m = 0; m < 4; ++m)
        #pragma unroll
        for (int n = 0; n < 2; ++n) {
            const size_t col = jbase + wn * 32 + n * 16 + ml;
            const int rbase = wm * 64 + m * 16 + half * 8;
            #pragma unroll
            for (int r = 0; r < 8; ++r)
                C[(size_t)(rbase + r) * GN + col] = acc[m][n][r];
        }
}

// ---------------- BN2 stats (z layout [oc][GN]) ----------------
__global__ __launch_bounds__(256) void bnstat2_kernel(const float* __restrict__ z,
                                                      const float* __restrict__ g,
                                                      const float* __restrict__ b,
                                                      float* __restrict__ scale,
                                                      float* __restrict__ shift) {
    const int c = blockIdx.x;
    const int tid = threadIdx.x;
    const float* row = z + (size_t)c * GN;
    float s = 0.f, ss = 0.f;
    for (int idx = tid; idx < GN; idx += 256) {
        float v = row[idx];
        s += v; ss += v * v;
    }
    __shared__ float rs[256], rq[256];
    rs[tid] = s; rq[tid] = ss;
    __syncthreads();
    for (int o = 128; o > 0; o >>= 1) {
        if (tid < o) { rs[tid] += rs[tid + o]; rq[tid] += rq[tid + o]; }
        __syncthreads();
    }
    if (tid == 0) {
        const float cnt = (float)GN;
        float m = rs[0] / cnt;
        float var = rq[0] / cnt - m * m;
        float rinv = rsqrtf(var + 1e-5f);
        scale[c] = g[c] * rinv;
        shift[c] = b[c] - m * g[c] * rinv;
    }
}

// ---------------- BN2 apply + ReLU + capsule transpose -> u[b][i][d] fp32 ----------------
__global__ __launch_bounds__(256) void bnapply2_kernel(const float* __restrict__ z,
                                                       const float* __restrict__ scale,
                                                       const float* __restrict__ shift,
                                                       float* __restrict__ u) {
    size_t i = (size_t)blockIdx.x * 256 + threadIdx.x;
    if (i >= (size_t)GM * GN) return;
    int oc = (int)(i / GN), j = (int)(i % GN);
    float v = fmaxf(fmaf(z[i], scale[oc], shift[oc]), 0.f);
    int n = j / 36, p = j % 36;
    int d = oc >> 5, gg = oc & 31;             // c = d*32 + gg ; i_caps = gg*36 + p
    u[((size_t)n * 1152 + (size_t)gg * 36 + p) * 8 + d] = v;
}

// ---------------- u_hat[b][j][i][e] = sum_d u[b][i][d] * W[j][i][d][e] ----------------
__global__ __launch_bounds__(256) void uhat_kernel(const float* __restrict__ u,
                                                   const float* __restrict__ W,
                                                   float* __restrict__ uh) {
    const int i = blockIdx.x;                    // 0..1151
    __shared__ float Ws[10 * 8 * 16];            // [j][d][e]
    __shared__ float us[NB * 8];                 // [b][d]
    for (int t = threadIdx.x; t < 1280; t += 256)
        Ws[t] = W[(size_t)(t / 128) * 147456 + (size_t)i * 128 + (t % 128)];
    for (int t = threadIdx.x; t < NB * 8; t += 256)
        us[t] = u[((size_t)(t >> 3) * 1152 + i) * 8 + (t & 7)];
    __syncthreads();
    for (int t = threadIdx.x; t < NB * 160; t += 256) {
        int e = t & 15, j = (t >> 4) % 10, b = t / 160;
        const float* wp = &Ws[(j * 8) * 16 + e];
        const float* up = &us[b * 8];
        float s = 0.f;
        #pragma unroll
        for (int d = 0; d < 8; ++d) s = fmaf(up[d], wp[d * 16], s);
        uh[(((size_t)b * 10 + j) * 1152 + i) * 16 + e] = s;
    }
}

// ---------------- zero b_log ----------------
__global__ __launch_bounds__(256) void zero_kernel(float* __restrict__ p, size_t n) {
    size_t i = (size_t)blockIdx.x * 256 + threadIdx.x;
    if (i < n) p[i] = 0.f;
}

// ---------------- routing: fused softmax + s + squash.  WG = (b,j). v -> d_out ----------
__global__ __launch_bounds__(256) void route_s_kernel(const float* __restrict__ uh,
                                                      const float* __restrict__ blog,
                                                      float* __restrict__ vout) {
    const int bj = blockIdx.x;                  // 0..5119
    const int b = bj / 10, j = bj % 10;
    const int tid = threadIdx.x;
    const float* uhb = uh + (size_t)b * 10 * 1152 * 16;
    const float* blb = blog + (size_t)b * 10 * 1152;
    float acc[16];
    #pragma unroll
    for (int e = 0; e < 16; ++e) acc[e] = 0.f;

    for (int i = tid; i < 1152; i += 256) {
        float bl[10], m = -3.0e38f;
        #pragma unroll
        for (int jj = 0; jj < 10; ++jj) {
            bl[jj] = blb[(size_t)jj * 1152 + i];
            m = fmaxf(m, bl[jj]);
        }
        float den = 0.f;
        #pragma unroll
        for (int jj = 0; jj < 10; ++jj) den += expf(bl[jj] - m);
        const float c = expf(bl[j] - m) / den;
        const float4* up = (const float4*)(uhb + ((size_t)j * 1152 + i) * 16);
        #pragma unroll
        for (int q = 0; q < 4; ++q) {
            float4 v4 = up[q];
            acc[4 * q + 0] = fmaf(c, v4.x, acc[4 * q + 0]);
            acc[4 * q + 1] = fmaf(c, v4.y, acc[4 * q + 1]);
            acc[4 * q + 2] = fmaf(c, v4.z, acc[4 * q + 2]);
            acc[4 * q + 3] = fmaf(c, v4.w, acc[4 * q + 3]);
        }
    }
    __shared__ float red[256][16];
    #pragma unroll
    for (int e = 0; e < 16; ++e) red[tid][e] = acc[e];
    __syncthreads();
    for (int o = 128; o > 0; o >>= 1) {
        if (tid < o)
            #pragma unroll
            for (int e = 0; e < 16; ++e) red[tid][e] += red[tid + o][e];
        __syncthreads();
    }
    if (tid == 0) {
        float nn = 0.f, s[16];
        #pragma unroll
        for (int e = 0; e < 16; ++e) { s[e] = red[0][e]; nn += s[e] * s[e]; }
        const float nrm = sqrtf(nn);
        const float f = nrm / (nn + 1.f);        // squash: n/(n^2+1) * s
        #pragma unroll
        for (int e = 0; e < 16; ++e) vout[(size_t)bj * 16 + e] = f * s[e];
    }
}

// ---------------- routing: b_log[b][j][i] += dot(u_hat[b][j][i][:], v[b][j][:]) -------
__global__ __launch_bounds__(256) void route_b_kernel(const float* __restrict__ uh,
                                                      const float* __restrict__ v,
                                                      float* __restrict__ blog) {
    size_t t = (size_t)blockIdx.x * 256 + threadIdx.x;      // b*10*1152 total
    if (t >= (size_t)NB * 10 * 1152) return;
    size_t bj = t / 1152;
    size_t i = t % 1152;
    const float4* up = (const float4*)(uh + bj * 1152 * 16 + i * 16);
    const float4* vp = (const float4*)(v + bj * 16);
    float s = 0.f;
    #pragma unroll
    for (int q = 0; q < 4; ++q) {
        float4 a = up[q], c = vp[q];
        s += a.x * c.x + a.y * c.y + a.z * c.z + a.w * c.w;
    }
    blog[t] += s;
}

extern "C" void kernel_launch(void* const* d_in, const int* in_sizes, int n_in,
                              void* d_out, int out_size, void* d_ws, size_t ws_size,
                              hipStream_t stream) {
    (void)in_sizes; (void)n_in; (void)out_size; (void)ws_size;
    const float* x  = (const float*)d_in[0];
    const float* w1 = (const float*)d_in[1];
    // conv1_b / conv2_b are mathematically cancelled by training-mode BN (mean subtraction)
    const float* g1 = (const float*)d_in[3];
    const float* b1 = (const float*)d_in[4];
    const float* w2 = (const float*)d_in[5];
    const float* g2 = (const float*)d_in[7];
    const float* b2 = (const float*)d_in[8];
    const float* W  = (const float*)d_in[9];
    float* vout = (float*)d_out;

    char* ws = (char*)d_ws;
    size_t off = 0;
    auto take = [&](size_t bytes) {
        char* p = ws + off;
        off += (bytes + 255) & ~(size_t)255;
        return p;
    };
    float*    y_raw = (float*)take((size_t)NB * GM * NPIX1 * 4);      // 209.7 MB
    _Float16* y_h   = (_Float16*)take((size_t)NB * GM * NPIX1 * 2);   // 104.9 MB
    float*    sc1   = (float*)take(4096);                              // scale|shift
    _Float16* A2    = (_Float16*)take((size_t)GM * GK * 2);           // 10.6 MB
    _Float16* B2    = (_Float16*)take((size_t)GN * GK * 2);           // 764.4 MB
    float*    z_raw = (float*)take((size_t)GM * GN * 4);              // 18.9 MB
    float*    sc2   = (float*)take(4096);
    float*    u     = (float*)take((size_t)NB * 1152 * 8 * 4);        // 18.9 MB
    float*    uh    = (float*)take((size_t)NB * 10 * 1152 * 16 * 4);  // 377.5 MB
    float*    blog  = (float*)take((size_t)NB * 10 * 1152 * 4);       // 23.6 MB

    packw2_kernel<<<(GM * GK + 255) / 256, 256, 0, stream>>>(w2, A2);
    conv1_kernel<<<NB * NPIX1, 256, 0, stream>>>(x, w1, y_raw);
    bnstat1_kernel<<<GM, 256, 0, stream>>>(y_raw, g1, b1, sc1, sc1 + 256);
    bnapply1_kernel<<<(NB * GM * NPIX1) / 256, 256, 0, stream>>>(y_raw, sc1, sc1 + 256, y_h);
    im2col_kernel<<<GN, 256, 0, stream>>>(y_h, B2);
    conv2_gemm_kernel<<<GN / 64, 256, 0, stream>>>(A2, B2, z_raw);
    bnstat2_kernel<<<GM, 256, 0, stream>>>(z_raw, g2, b2, sc2, sc2 + 256);
    bnapply2_kernel<<<(GM * GN) / 256, 256, 0, stream>>>(z_raw, sc2, sc2 + 256, u);
    uhat_kernel<<<1152, 256, 0, stream>>>(u, W, uh);
    zero_kernel<<<(NB * 10 * 1152) / 256, 256, 0, stream>>>(blog, (size_t)NB * 10 * 1152);
    for (int t = 0; t < 5; ++t) {
        route_s_kernel<<<NB * 10, 256, 0, stream>>>(uh, blog, vout);
        if (t < 4)
            route_b_kernel<<<(NB * 10 * 1152) / 256, 256, 0, stream>>>(uh, vout, blog);
    }
}
